// PairStack_89996744721176
// MI455X (gfx1250) — compile-verified, matrix operand order checked by reading, not run on hardware
//
#include <hip/hip_runtime.h>

// ---------------------------------------------------------------------------
// PairStack for MI455X (gfx1250, wave32, WMMA f32_16x16x32_f16)
// B=2, N=192, D=128, H=4, C=32, N_PT=4.  All GEMM-shaped math on WMMA
// (incl. the 4-col bias head, padded to a 16-col tile), LN/softmax/sigmoid
// on f32 VALU with fast rcp.  Whole working set is L2-resident.
// ---------------------------------------------------------------------------

#define Bb 2
#define Nn 192
#define Dd 128
#define Cc 32
#define Hh 4
#define TOK (Bb * Nn * Nn)      // 73728 pair tokens
#define LN_EPS 1e-5f
#define NCOLS 528               // 384 qkv + 128 gate + 16 bias-head (padded)

typedef __attribute__((ext_vector_type(16))) _Float16 v16h;
typedef __attribute__((ext_vector_type(8)))  _Float16 v8h;
typedef __attribute__((ext_vector_type(8)))  float    v8f;

__device__ __forceinline__ v8f wmma16(v16h a, v16h b, v8f c) {
  return __builtin_amdgcn_wmma_f32_16x16x32_f16(false, a, false, b,
                                                (short)0, c, false, false);
}

// A-type (and B-from-B^T) fragment load: tile is row-major [16][>=32] f16.
__device__ __forceinline__ v16h ld_frag(const _Float16* base, int ld) {
  int lane = threadIdx.x & 31;
  const _Float16* p = base + (size_t)(lane & 15) * ld + ((lane >> 4) << 3);
  v8h lo = *(const v8h*)p;
  v8h hi = *(const v8h*)(p + 16);
  v16h f;
#pragma unroll
  for (int i = 0; i < 8; ++i) { f[i] = lo[i]; f[i + 8] = hi[i]; }
  return f;
}

// B-fragment from a K-major [nrow][ld] tile without pre-transpose.
__device__ __forceinline__ v16h ld_frag_T(const _Float16* vb, int ld, int k0, int c0) {
  int lane = threadIdx.x & 31;
  int c  = c0 + (lane & 15);
  int kh = k0 + ((lane >> 4) << 3);
  v16h f;
#pragma unroll
  for (int i = 0; i < 8; ++i) {
    f[i]     = vb[(kh + i) * ld + c];
    f[i + 8] = vb[(kh + 16 + i) * ld + c];
  }
  return f;
}

__device__ __forceinline__ float fast_sigmoid(float x) {
  return __builtin_amdgcn_rcpf(1.f + __expf(-x));   // v_exp + v_rcp only
}

// ---------------------------------------------------------------------------
// Weight transpose + f32->f16 convert:  W[K][M] -> WT[M][K]
// ---------------------------------------------------------------------------
__global__ void k_transpose_f16(const float* __restrict__ W,
                                _Float16* __restrict__ WT, int K, int M) {
  int idx = blockIdx.x * 256 + threadIdx.x;
  if (idx >= K * M) return;
  int m = idx / K, k = idx % K;
  WT[(size_t)m * K + k] = (_Float16)W[(size_t)k * M + m];
}

// Wbias (128,4) -> 16x128 padded transpose (rows 4..15 zero)
__global__ void k_prep_biasT(const float* __restrict__ Wbias,
                             _Float16* __restrict__ WT) {
  int idx = blockIdx.x * 256 + threadIdx.x;
  if (idx >= 16 * Dd) return;
  int m = idx >> 7, k = idx & 127;
  WT[idx] = (_Float16)(m < Hh ? Wbias[k * Hh + m] : 0.f);
}

// ---------------------------------------------------------------------------
// K1: LayerNorm + fused qkv/gate/bias projection.
// 64 tokens per block, 256 threads (8 waves); 33 col-tiles round-robined.
// Combined WT layout [528][128]: 0..383 qkv, 384..511 gate, 512..527 bias.
// ---------------------------------------------------------------------------
__global__ __launch_bounds__(256) void k_ln_qkvg(
    const float* __restrict__ z, const float* __restrict__ ln_g,
    const float* __restrict__ ln_b, const _Float16* __restrict__ WT,
    const float* __restrict__ Wg_b,
    _Float16* __restrict__ qbuf, _Float16* __restrict__ kbuf,
    _Float16* __restrict__ vbuf, _Float16* __restrict__ gate,
    float* __restrict__ biasb) {
  __shared__ __align__(16) _Float16 zn[64][Dd];
  const int t0  = blockIdx.x * 64;
  const int tid = threadIdx.x;
  // t0 is a multiple of 64 and N*N%64==0 -> batch index uniform per block
  const int bblk = t0 / (Nn * Nn);
  const int ij0  = t0 - bblk * (Nn * Nn);

  // ---- LayerNorm: 4 lanes per token, 32 channels each ----
  {
    int seg = (tid & 3) * 32;
    const float* zp = z + (size_t)(t0 + (tid >> 2)) * Dd + seg;
    float vals[32], s = 0.f, s2 = 0.f;
#pragma unroll
    for (int i = 0; i < 32; ++i) { float x = zp[i]; vals[i] = x; s += x; s2 += x * x; }
    for (int off = 1; off < 4; off <<= 1) {
      s += __shfl_xor(s, off, 32); s2 += __shfl_xor(s2, off, 32);
    }
    float mu  = s * (1.f / 128.f);
    float var = s2 * (1.f / 128.f) - mu * mu;
    float rs  = rsqrtf(var + LN_EPS);
#pragma unroll
    for (int i = 0; i < 32; ++i) {
      float xn = (vals[i] - mu) * rs * ln_g[seg + i] + ln_b[seg + i];
      zn[tid >> 2][seg + i] = (_Float16)xn;
    }
  }
  __syncthreads();

  // ---- GEMM: [64x128] x [128x528] ----
  const int wave = tid >> 5, lane = tid & 31;
  const int cn = lane & 15, rb = (lane >> 4) << 3;
  const float qscale = 0.4204482076268573f;  // 32^(-1/4)

  for (int nt = wave; nt < NCOLS / 16; nt += 8) {
    const int col = __builtin_amdgcn_readfirstlane(nt * 16);  // wave-uniform
    const _Float16* wb = WT + (size_t)col * Dd;
    for (int mf = 0; mf < 4; ++mf) {
      v8f acc = {};
#pragma unroll
      for (int kk = 0; kk < 4; ++kk)
        acc = wmma16(ld_frag(&zn[mf * 16][kk * 32], Dd),
                     ld_frag(wb + kk * 32, Dd), acc);
      const int row0 = mf * 16 + rb;    // first of 8 rows for this lane
      if (col < 384) {                  // q / k / v  (scalar branch)
        int part = col >> 7;
        int h = (col >> 5) & 3;
        _Float16* dbuf = part == 0 ? qbuf : part == 1 ? kbuf : vbuf;
        float scl = (part == 2) ? 1.f : qscale;
        size_t base = ((size_t)(bblk * Hh + h) * Nn * Nn + (ij0 + row0)) * Cc
                      + (col & 31) + cn;
#pragma unroll
        for (int r = 0; r < 8; ++r)
          dbuf[base + (size_t)r * Cc] = (_Float16)(acc[r] * scl);
      } else if (col < 512) {           // sigmoid gate
        int gm = col - 384 + cn;
        float wbv = Wg_b[gm];
        size_t base = (size_t)(t0 + row0) * Dd + gm;
#pragma unroll
        for (int r = 0; r < 8; ++r)
          gate[base + (size_t)r * Dd] = (_Float16)fast_sigmoid(acc[r] + wbv);
      } else if (cn < Hh) {             // bias head (cols 512..515 valid)
        size_t base = (size_t)(t0 + row0) * Hh + cn;
#pragma unroll
        for (int r = 0; r < 8; ++r)
          biasb[base + (size_t)r * Hh] = acc[r];
      }
    }
  }
}

// ---------------------------------------------------------------------------
// K2: attention, one block per (b,h,fixed). mode 0 = 'start', 1 = 'end'.
// 128 threads = 4 waves x 16 queries x 3 iterations.
// ---------------------------------------------------------------------------
__global__ __launch_bounds__(128) void k_attn(
    const _Float16* __restrict__ qbuf, const _Float16* __restrict__ kbuf,
    const _Float16* __restrict__ vbuf, const float* __restrict__ biasb,
    _Float16* __restrict__ obuf, int mode) {
  __shared__ __align__(16) _Float16 Kt[Nn * Cc];      // 12 KB
  __shared__ __align__(16) _Float16 Vt[Nn * Cc];      // 12 KB
  __shared__ __align__(16) _Float16 Pr[4][16 * Nn];   // 24 KB probs

  const int u = blockIdx.x;
  const int b = u / (Hh * Nn);
  const int rem = u % (Hh * Nn);
  const int h = rem / Nn, fix = rem % Nn;

  size_t tb; int rowstride;
  long bbase, bqs, bks;
  if (mode == 0) {                       // start: rows = j, keys = k, same i
    tb = ((size_t)(b * Hh + h) * Nn + fix) * (size_t)Nn * Cc;
    rowstride = Cc;
    bbase = (long)b * Nn * Nn * Hh + h; bqs = (long)Nn * Hh; bks = Hh;
  } else {                               // end: rows = i, keys = k, same j
    tb = ((size_t)(b * Hh + h) * Nn * Nn + fix) * Cc;
    rowstride = Nn * Cc;
    bbase = (long)b * Nn * Nn * Hh + (long)fix * Hh + h; bqs = 0; bks = (long)Nn * Hh;
  }

  const int tid = threadIdx.x;
  for (int idx = tid; idx < Nn * Cc; idx += 128) {
    int r = idx >> 5, c = idx & 31;
    Kt[idx] = kbuf[tb + (size_t)r * rowstride + c];
    Vt[idx] = vbuf[tb + (size_t)r * rowstride + c];
  }
  __syncthreads();

  const int wave = tid >> 5, lane = tid & 31;
  const int cn = lane & 15, rb = (lane >> 4) << 3;
  _Float16* pw = &Pr[wave][0];

  for (int qi = 0; qi < 3; ++qi) {
    int q0 = (qi * 4 + wave) * 16;
    v16h qf = ld_frag(qbuf + tb + (size_t)q0 * rowstride, rowstride);

    v8f sc[12];
#pragma unroll
    for (int kf = 0; kf < 12; ++kf) {
      v8f zr = {};
      sc[kf] = wmma16(qf, ld_frag(Kt + kf * 16 * Cc, Cc), zr);
    }
#pragma unroll
    for (int kf = 0; kf < 12; ++kf) {
      int kc = kf * 16 + cn;
#pragma unroll
      for (int r = 0; r < 8; ++r)
        sc[kf][r] += biasb[bbase + (long)(q0 + r + rb) * bqs + (long)kc * bks];
    }
    // softmax across 192 keys (row lives in 16 lanes x 12 frags)
    float li[8];
#pragma unroll
    for (int r = 0; r < 8; ++r) {
      float m = sc[0][r];
#pragma unroll
      for (int kf = 1; kf < 12; ++kf) m = fmaxf(m, sc[kf][r]);
      for (int off = 1; off < 16; off <<= 1) m = fmaxf(m, __shfl_xor(m, off, 32));
      float sum = 0.f;
#pragma unroll
      for (int kf = 0; kf < 12; ++kf) {
        float e = __expf(sc[kf][r] - m);
        sc[kf][r] = e; sum += e;
      }
      for (int off = 1; off < 16; off <<= 1) sum += __shfl_xor(sum, off, 32);
      li[r] = __builtin_amdgcn_rcpf(sum);
    }
    // spill probs as f16 A-operand (per-wave slice, LDS in-order within wave)
#pragma unroll
    for (int kf = 0; kf < 12; ++kf)
#pragma unroll
      for (int r = 0; r < 8; ++r)
        pw[(r + rb) * Nn + kf * 16 + cn] = (_Float16)sc[kf][r];

    // o = P(16x192) x V(192x32)
#pragma unroll
    for (int cf = 0; cf < 2; ++cf) {
      v8f oc = {};
#pragma unroll
      for (int ks = 0; ks < 6; ++ks)
        oc = wmma16(ld_frag(pw + ks * 32, Nn),
                    ld_frag_T(Vt, Cc, ks * 32, cf * 16), oc);
#pragma unroll
      for (int r = 0; r < 8; ++r) {
        int qr = q0 + r + rb;
        obuf[tb + (size_t)qr * rowstride + cf * 16 + cn] =
            (_Float16)(oc[r] * li[r]);
      }
    }
  }
}

// ---------------------------------------------------------------------------
// K3: x = gate * o (re-interleaved (c,h)), out = x @ Wo^T + b + residual
// ---------------------------------------------------------------------------
__global__ __launch_bounds__(256) void k_gate_out(
    const _Float16* __restrict__ obuf, const _Float16* __restrict__ gate,
    const _Float16* __restrict__ WoT, const float* __restrict__ Wo_b,
    const float* __restrict__ zres, float* __restrict__ zout) {
  __shared__ __align__(16) _Float16 xs[64][Dd];
  const int t0 = blockIdx.x * 64;
  const int tid = threadIdx.x;
  const int bblk = t0 / (Nn * Nn);           // uniform per block
  const int ij0  = t0 - bblk * (Nn * Nn);
  {
    int trow = tid >> 2;
    int ij = ij0 + trow;
    int seg = (tid & 3) * 32;
#pragma unroll
    for (int i = 0; i < 32; ++i) {
      int m = seg + i, c = m >> 2, hh = m & 3;
      size_t oi = ((size_t)(bblk * Hh + hh) * Nn * Nn + ij) * Cc + c;
      float x = (float)obuf[oi] * (float)gate[(size_t)(t0 + trow) * Dd + m];
      xs[trow][m] = (_Float16)x;
    }
  }
  __syncthreads();

  const int wave = tid >> 5, lane = tid & 31;
  const int col0 = wave * 16;
  const int cn = lane & 15, rb = (lane >> 4) << 3;
  for (int mf = 0; mf < 4; ++mf) {
    v8f acc = {};
#pragma unroll
    for (int kk = 0; kk < 4; ++kk)
      acc = wmma16(ld_frag(&xs[mf * 16][kk * 32], Dd),
                   ld_frag(WoT + (size_t)col0 * Dd + kk * 32, Dd), acc);
    int m = col0 + cn;
    size_t base = (size_t)(t0 + mf * 16 + rb) * Dd + m;
    float wob = Wo_b[m];
#pragma unroll
    for (int r = 0; r < 8; ++r)
      zout[base + (size_t)r * Dd] = acc[r] + wob + zres[base + (size_t)r * Dd];
  }
}

// ---------------------------------------------------------------------------
// K4: pair transition part 1: a = relu(LN(z) @ Wa^T + b)  (512 cols)
// ---------------------------------------------------------------------------
__global__ __launch_bounds__(256) void k_pt_a(
    const float* __restrict__ z, const float* __restrict__ ln_g,
    const float* __restrict__ ln_b, const _Float16* __restrict__ WaT,
    const float* __restrict__ Wa_b, _Float16* __restrict__ abuf) {
  __shared__ __align__(16) _Float16 zn[64][Dd];
  const int t0 = blockIdx.x * 64;
  const int tid = threadIdx.x;
  {
    int seg = (tid & 3) * 32;
    const float* zp = z + (size_t)(t0 + (tid >> 2)) * Dd + seg;
    float vals[32], s = 0.f, s2 = 0.f;
#pragma unroll
    for (int i = 0; i < 32; ++i) { float x = zp[i]; vals[i] = x; s += x; s2 += x * x; }
    for (int off = 1; off < 4; off <<= 1) {
      s += __shfl_xor(s, off, 32); s2 += __shfl_xor(s2, off, 32);
    }
    float mu = s * (1.f / 128.f);
    float var = s2 * (1.f / 128.f) - mu * mu;
    float rs = rsqrtf(var + LN_EPS);
#pragma unroll
    for (int i = 0; i < 32; ++i)
      zn[tid >> 2][seg + i] = (_Float16)((vals[i] - mu) * rs * ln_g[seg + i] + ln_b[seg + i]);
  }
  __syncthreads();

  const int wave = tid >> 5, lane = tid & 31;
  const int cn = lane & 15, rb = (lane >> 4) << 3;
  for (int nt = wave; nt < 32; nt += 8) {
    const int col = __builtin_amdgcn_readfirstlane(nt * 16);
    const _Float16* wb = WaT + (size_t)col * Dd;
    float wab = Wa_b[col + cn];
    for (int mf = 0; mf < 4; ++mf) {
      v8f acc = {};
#pragma unroll
      for (int kk = 0; kk < 4; ++kk)
        acc = wmma16(ld_frag(&zn[mf * 16][kk * 32], Dd),
                     ld_frag(wb + kk * 32, Dd), acc);
      size_t base = (size_t)(t0 + mf * 16 + rb) * 512 + col + cn;
#pragma unroll
      for (int r = 0; r < 8; ++r)
        abuf[base + (size_t)r * 512] = (_Float16)fmaxf(acc[r] + wab, 0.f);
    }
  }
}

// ---------------------------------------------------------------------------
// K5: pair transition part 2: out = a @ Wo^T + b + residual  (K=512)
// ---------------------------------------------------------------------------
__global__ __launch_bounds__(256) void k_pt_out(
    const _Float16* __restrict__ abuf, const _Float16* __restrict__ WoT2,
    const float* __restrict__ Wo_b, const float* __restrict__ zres,
    float* __restrict__ zout) {
  const int t0 = blockIdx.x * 64;
  const int tid = threadIdx.x;
  const int wave = tid >> 5, lane = tid & 31;
  const int col0 = wave * 16;
  const int cn = lane & 15, rb = (lane >> 4) << 3;
  for (int mf = 0; mf < 4; ++mf) {
    v8f acc = {};
    const _Float16* arow = abuf + (size_t)(t0 + mf * 16) * 512;
#pragma unroll
    for (int kk = 0; kk < 16; ++kk)
      acc = wmma16(ld_frag(arow + kk * 32, 512),
                   ld_frag(WoT2 + (size_t)col0 * 512 + kk * 32, 512), acc);
    int m = col0 + cn;
    size_t base = (size_t)(t0 + mf * 16 + rb) * Dd + m;
    float wob = Wo_b[m];
#pragma unroll
    for (int r = 0; r < 8; ++r)
      zout[base + (size_t)r * Dd] = acc[r] + wob + zres[base + (size_t)r * Dd];
  }
}

// ---------------------------------------------------------------------------
// Host launcher
// ---------------------------------------------------------------------------
extern "C" void kernel_launch(void* const* d_in, const int* in_sizes, int n_in,
                              void* d_out, int out_size, void* d_ws, size_t ws_size,
                              hipStream_t stream) {
  const float* zij   = (const float*)d_in[0];
  const float* s_ln_g = (const float*)d_in[1];
  const float* s_ln_b = (const float*)d_in[2];
  const float* s_Wqkv = (const float*)d_in[3];
  const float* s_Wbias= (const float*)d_in[4];
  const float* s_Wg   = (const float*)d_in[5];
  const float* s_Wg_b = (const float*)d_in[6];
  const float* s_Wo   = (const float*)d_in[7];
  const float* s_Wo_b = (const float*)d_in[8];
  const float* e_ln_g = (const float*)d_in[9];
  const float* e_ln_b = (const float*)d_in[10];
  const float* e_Wqkv = (const float*)d_in[11];
  const float* e_Wbias= (const float*)d_in[12];
  const float* e_Wg   = (const float*)d_in[13];
  const float* e_Wg_b = (const float*)d_in[14];
  const float* e_Wo   = (const float*)d_in[15];
  const float* e_Wo_b = (const float*)d_in[16];
  const float* p_ln_g = (const float*)d_in[17];
  const float* p_ln_b = (const float*)d_in[18];
  const float* p_Wa   = (const float*)d_in[19];
  const float* p_Wa_b = (const float*)d_in[20];
  const float* p_Wo   = (const float*)d_in[21];
  const float* p_Wo_b = (const float*)d_in[22];

  float* zout = (float*)d_out;

  // ---- workspace layout ----
  char* ws = (char*)d_ws;
  const size_t TD = (size_t)TOK * Dd;
  _Float16* qbuf = (_Float16*)ws;            // [B][H][N][N][C]
  _Float16* kbuf = qbuf + TD;
  _Float16* vbuf = kbuf + TD;
  _Float16* gate = vbuf + TD;                // [token][128], sigmoid applied
  _Float16* obuf = gate + TD;                // [B][H][N][N][C]
  float*    biasb = (float*)(obuf + TD);     // [token][H] f32
  _Float16* wreg  = (_Float16*)(biasb + (size_t)TOK * Hh);
  _Float16* WT_s  = wreg;                    // [528][128] combined qkv|gate|bias
  _Float16* WoT_s = WT_s + (size_t)NCOLS * Dd;
  _Float16* WT_e  = WoT_s + 128 * 128;
  _Float16* WoT_e = WT_e + (size_t)NCOLS * Dd;
  _Float16* WaT   = WoT_e + 128 * 128;       // [512][128]
  _Float16* WoT2  = WaT + 512 * 128;         // [128][512]
  _Float16* abuf  = qbuf;                    // PT phase reuses q..o region

  auto tr = [&](const float* W, _Float16* WT, int K, int M) {
    int n = K * M;
    k_transpose_f16<<<(n + 255) / 256, 256, 0, stream>>>(W, WT, K, M);
  };
  // combined WT per TA layer: rows 0..383 qkv^T, 384..511 Wg^T, 512..527 bias^T
  tr(s_Wqkv, WT_s, 128, 384);
  tr(s_Wg,   WT_s + 384 * Dd, 128, 128);
  k_prep_biasT<<<(16 * Dd + 255) / 256, 256, 0, stream>>>(s_Wbias, WT_s + 512 * Dd);
  tr(s_Wo,   WoT_s, 128, 128);
  tr(e_Wqkv, WT_e, 128, 384);
  tr(e_Wg,   WT_e + 384 * Dd, 128, 128);
  k_prep_biasT<<<(16 * Dd + 255) / 256, 256, 0, stream>>>(e_Wbias, WT_e + 512 * Dd);
  tr(e_Wo,   WoT_e, 128, 128);
  tr(p_Wa,   WaT,  128, 512);
  tr(p_Wo,   WoT2, 512, 128);

  const int nTiles = TOK / 64;       // 1152
  const int nAttn  = Bb * Hh * Nn;   // 1536

  // ---- layer 1: triangle attention 'start' ----
  k_ln_qkvg<<<nTiles, 256, 0, stream>>>(zij, s_ln_g, s_ln_b, WT_s, s_Wg_b,
                                        qbuf, kbuf, vbuf, gate, biasb);
  k_attn<<<nAttn, 128, 0, stream>>>(qbuf, kbuf, vbuf, biasb, obuf, 0);
  k_gate_out<<<nTiles, 256, 0, stream>>>(obuf, gate, WoT_s, s_Wo_b, zij, zout);

  // ---- layer 2: triangle attention 'end' ----
  k_ln_qkvg<<<nTiles, 256, 0, stream>>>(zout, e_ln_g, e_ln_b, WT_e, e_Wg_b,
                                        qbuf, kbuf, vbuf, gate, biasb);
  k_attn<<<nAttn, 128, 0, stream>>>(qbuf, kbuf, vbuf, biasb, obuf, 1);
  k_gate_out<<<nTiles, 256, 0, stream>>>(obuf, gate, WoT_e, e_Wo_b, zout, zout);

  // ---- pair transition ----
  k_pt_a<<<nTiles, 256, 0, stream>>>(zout, p_ln_g, p_ln_b, WaT, p_Wa_b, abuf);
  k_pt_out<<<nTiles, 256, 0, stream>>>(abuf, WoT2, p_Wo_b, zout, zout);
}
